// ColorRestoration_20005957665584
// MI455X (gfx1250) — matrix operand order repaired and checked by reading, not compiled, not run on hardware
//
#include <hip/hip_runtime.h>
#include <hip/hip_bf16.h>
#include <math.h>

// ---------------- problem constants (match reference) ----------------
#define BB      8
#define HH      128
#define WWD     128
#define HWP     (HH*WWD)        // 16384 pixels
#define PW      130             // padded width/height (1-px zero halo)
#define PHW     (PW*PW)         // 16900 padded pixels
#define CHN     128
#define BASEC   64
#define TED     256
#define TSTEPS  128
#define GNEPS   1e-5f

typedef __attribute__((ext_vector_type(16))) __bf16 v16bf;
typedef __attribute__((ext_vector_type(8)))  float  v8f;

union FragBF { v16bf v; uint4 q[2]; unsigned int u[8]; };

__device__ __forceinline__ unsigned short f2bf(float f) {
  union { float f; unsigned u; } x; x.f = f;
  unsigned u = x.u;
  unsigned r = 0x7FFFu + ((u >> 16) & 1u);   // round-to-nearest-even
  return (unsigned short)((u + r) >> 16);
}
__device__ __forceinline__ float siluf(float y) { return y / (1.f + __expf(-y)); }
__device__ __forceinline__ float sigf(float y)  { return 1.f / (1.f + __expf(-y)); }

// =====================================================================
// Implicit-GEMM 3x3 SAME conv via v_wmma_f32_16x16x32_bf16.
// actB : bf16 channels-last, zero-padded halo: [B][130*130][CIN]
// wB   : bf16 [sel][Cout][9][CIN]  (tap-major, CIN contiguous)
// Per wave: 16 out-channels x 16 pixels; K = 9 taps * CIN, 32 per WMMA.
// Operand fragments are two contiguous 16-byte runs each -> b128 loads:
//   A (16x32 bf16): lane m=lane&15; K run at half*8 and 16+half*8
//   B (32x16 bf16): lane n=lane&15; K run at half*8 and 16+half*8
// MODE 0: out32[b][p][Cout] = acc + bias (channels-last, unpadded)
// MODE 1: sigmoid, scatter first 3 channels to NCHW output (restored)
// =====================================================================
template <int CIN, int MODE>
__global__ void conv3x3_wmma_k(const unsigned short* __restrict__ actB,
                               const unsigned short* __restrict__ wB,
                               const float* __restrict__ bias, int biasStride,
                               const int* __restrict__ tsel,
                               float* __restrict__ out32,
                               float* __restrict__ outNCHW,
                               int Cout)
{
  const int lane = threadIdx.x & 31;
  const int wv   = threadIdx.x >> 5;          // out-channel tile of this wave
  const int b    = blockIdx.y;
  const int half = lane >> 4;
  const int mrow = lane & 15;
  const int widx = tsel ? tsel[b] : 0;
  const unsigned short* wrow =
      wB + ((size_t)widx * Cout + wv * 16 + mrow) * (9 * CIN) + half * 8;
  const int p  = blockIdx.x * 16 + mrow;      // this lane's pixel (B operand N, D N)
  const int py = p >> 7, px = p & 127;
  const unsigned short* actb =
      actB + (size_t)b * PHW * CIN + half * 8;   // B-operand K base folded in

  v8f acc = {};
#pragma unroll
  for (int tap = 0; tap < 9; ++tap) {
    const int dy = tap / 3, dx = tap - dy * 3;
    // halo layout: src pixel (py+dy-1, px+dx-1) -> padded row py+dy, col px+dx
    const unsigned short* ap = actb + (size_t)((py + dy) * PW + (px + dx)) * CIN;
    const unsigned short* wp = wrow + tap * CIN;
#pragma unroll
    for (int kc = 0; kc < CIN; kc += 32) {
      FragBF a, bm;
      a.q[0]  = *(const uint4*)(wp + kc);
      a.q[1]  = *(const uint4*)(wp + kc + 16);
      bm.q[0] = *(const uint4*)(ap + kc);
      bm.q[1] = *(const uint4*)(ap + kc + 16);
      acc = __builtin_amdgcn_wmma_f32_16x16x32_bf16(false, a.v, false, bm.v,
                                                    (short)0, acc, false, false);
    }
  }
  // D layout: VGPR r -> M = r (lanes0-15) / 8+r (lanes16-31); N = lane&15
  const int cb = wv * 16 + half * 8;
  if (MODE == 0) {
    float* o = out32 + ((size_t)b * HWP + p) * Cout + cb;
    const float* bp = bias + (size_t)widx * biasStride + cb;
#pragma unroll
    for (int r = 0; r < 8; ++r) o[r] = acc[r] + bp[r];
  } else {
#pragma unroll
    for (int r = 0; r < 8; ++r) {
      const int ch = cb + r;
      if (ch < 3) {
        const float y = acc[r] + bias[ch];
        outNCHW[((size_t)b * 3 + ch) * HWP + p] = sigf(y);
      }
    }
  }
}

// -------- zero a buffer with b128 stores (halo init, graph-safe) -----
__global__ void zero_u4_k(uint4* __restrict__ p, unsigned n16) {
  const unsigned i = blockIdx.x * 256 + threadIdx.x;
  if (i < n16) p[i] = make_uint4(0u, 0u, 0u, 0u);
}

// -------- feature-extraction conv1: 3->64, VALU (27 MACs/out) --------
__global__ void fe1_k(const float* __restrict__ x, const float* __restrict__ w,
                      const float* __restrict__ bias, float* __restrict__ out) {
  const int tid = threadIdx.x;
  const int pix = blockIdx.x * 4 + (tid >> 6);
  const int o   = tid & 63;
  const int b   = pix / HWP, p = pix % HWP;
  const int py = p >> 7, px = p & 127;
  float s = bias[o];
  for (int ci = 0; ci < 3; ++ci) {
    const float* xp = x + ((size_t)(b * 3 + ci)) * HWP;
    const float* wp = w + (o * 3 + ci) * 9;
    for (int tap = 0; tap < 9; ++tap) {
      const int dy = tap / 3, dx = tap - dy * 3;
      const int sy = py + dy - 1, sx = px + dx - 1;
      if (sy >= 0 && sy < HH && sx >= 0 && sx < WWD) s += xp[sy * WWD + sx] * wp[tap];
    }
  }
  out[((size_t)b * HWP + p) * BASEC + o] = s;
}

// -------- GroupNorm stats (groups=8), channels-last, LDS reduce ------
__global__ void gn_stats_k(const float* __restrict__ src, int C, int cpg,
                           float* __restrict__ stats) {
  const int bg = blockIdx.x, b = bg >> 3, g = bg & 7;
  const int n = HWP * cpg;
  float s = 0.f, q = 0.f;
  for (int i = threadIdx.x; i < n; i += 256) {
    const int p = i / cpg, c = g * cpg + (i - p * cpg);
    const float v = src[((size_t)b * HWP + p) * C + c];
    s += v; q += v * v;
  }
  __shared__ float ss[256], sq[256];
  ss[threadIdx.x] = s; sq[threadIdx.x] = q; __syncthreads();
  for (int o = 128; o > 0; o >>= 1) {
    if (threadIdx.x < (unsigned)o) { ss[threadIdx.x] += ss[threadIdx.x + o]; sq[threadIdx.x] += sq[threadIdx.x + o]; }
    __syncthreads();
  }
  if (threadIdx.x == 0) {
    const float m = ss[0] / n, var = sq[0] / n - m * m;
    stats[bg * 2] = m; stats[bg * 2 + 1] = rsqrtf(var + GNEPS);
  }
}

// -------- GN apply + SiLU; emit bf16 (padded halo layout) / fp32 -----
__global__ void gn_apply_k(const float* __restrict__ src, int C, int cpg,
                           const float* __restrict__ stats,
                           const float* __restrict__ scale, const float* __restrict__ bias,
                           const int* __restrict__ tsel, int selStride,
                           unsigned short* __restrict__ outB, float* __restrict__ outF) {
  const size_t idx = (size_t)blockIdx.x * 256 + threadIdx.x;
  const size_t total = (size_t)BB * HWP * C;
  if (idx >= total) return;
  const int c = (int)(idx % C);
  const int p = (int)((idx / C) % HWP);
  const int b = (int)(idx / ((size_t)HWP * C));
  const int g = c / cpg;
  const float m = stats[(b * 8 + g) * 2], rs = stats[(b * 8 + g) * 2 + 1];
  const int w = tsel ? tsel[b] * selStride : 0;
  float y = (src[idx] - m) * rs * scale[w + c] + bias[w + c];
  y = siluf(y);
  if (outF) outF[idx] = y;
  if (outB) {
    const int py = p >> 7, px = p & 127;
    outB[((size_t)b * PHW + (py + 1) * PW + (px + 1)) * C + c] = f2bf(y);
  }
}

// -------- stage shared fp32 weights [O][I][3][3] -> bf16 [Opad][9][I] -
__global__ void stage_w_k(const float* __restrict__ w, unsigned short* __restrict__ wB,
                          int O, int I, int Opad) {
  const int idx = blockIdx.x * 256 + threadIdx.x;
  if (idx >= Opad * 9 * I) return;
  const int i = idx % I, tap = (idx / I) % 9, o = idx / (9 * I);
  wB[idx] = (o < O) ? f2bf(w[((size_t)o * I + i) * 9 + tap]) : (unsigned short)0;
}

// -------- stage gathered expert weights -> bf16 [b][128][9][128] -----
__global__ void stage_we_k(const float* __restrict__ w, const int* __restrict__ tsel,
                           unsigned short* __restrict__ wB, int Isrc) {
  const int idx = blockIdx.x * 256 + threadIdx.x;
  const int i = idx & 127, tap = (idx >> 7) % 9, o = (idx / 1152) & 127, b = idx / 147456;
  const float* s = w + (size_t)tsel[b] * 128 * Isrc * 9;
  wB[idx] = f2bf(s[((size_t)o * Isrc + i) * 9 + tap]);
}

// -------- time MLP: temb -> silu(w1@te+b1) -> mean(w2@h+b2) = tc -----
__global__ void mlp_tc_k(const int* __restrict__ t, const float* __restrict__ w1,
                         const float* __restrict__ b1, const float* __restrict__ w2,
                         const float* __restrict__ b2, float* __restrict__ tc) {
  const int b = blockIdx.x, tid = threadIdx.x;
  const int ti = t[b];
  __shared__ float te[TED], h[TED], red[128];
  const float step = -logf(10000.f) / 127.f;
  const int i = tid & 127;
  const float e = (float)ti * __expf(step * (float)i);
  te[tid] = (tid < 128) ? __sinf(e) : __cosf(e);
  __syncthreads();
  const float* w1r = w1 + ((size_t)ti * TED + tid) * TED;
  float s = b1[ti * TED + tid];
  for (int k = 0; k < TED; ++k) s += w1r[k] * te[k];
  h[tid] = siluf(s);
  __syncthreads();
  if (tid < 128) {
    const float* w2r = w2 + ((size_t)ti * CHN + tid) * TED;
    float s2 = b2[ti * CHN + tid];
    for (int k = 0; k < TED; ++k) s2 += w2r[k] * h[k];
    red[tid] = s2;
  }
  __syncthreads();
  for (int o = 64; o > 0; o >>= 1) { if (tid < o) red[tid] += red[tid + o]; __syncthreads(); }
  if (tid == 0) tc[b] = red[0] / 128.f;
}

// -------- add 129th-channel (constant time map) conv contribution ----
__global__ void tadd_k(float* __restrict__ z, const float* __restrict__ c1w,
                       const int* __restrict__ tsel, const float* __restrict__ tc) {
  const size_t idx = (size_t)blockIdx.x * 256 + threadIdx.x;
  const int c = (int)(idx & 127);
  const int p = (int)((idx >> 7) & (HWP - 1));
  const int b = (int)(idx / ((size_t)HWP * 128));
  const int py = p >> 7, px = p & 127;
  const float* wrow = c1w + (((size_t)tsel[b] * 128 + c) * 129 + 128) * 9;
  float s = 0.f;
  for (int tap = 0; tap < 9; ++tap) {
    const int dy = tap / 3, dx = tap - dy * 3;
    const int sy = py + dy - 1, sx = px + dx - 1;
    if (sy >= 0 && sy < HH && sx >= 0 && sx < WWD) s += wrow[tap];
  }
  z[idx] += s * tc[b];
}

// -------- x_t = feat - alpha*noise ; fp32 in-place + bf16 (padded) ---
__global__ void xt_k(float* __restrict__ feat32, const float* __restrict__ noise,
                     const int* __restrict__ tsel, unsigned short* __restrict__ xtB) {
  const size_t idx = (size_t)blockIdx.x * 256 + threadIdx.x;
  const int c = (int)(idx & 127);
  const int p = (int)((idx >> 7) & (HWP - 1));
  const int b = (int)(idx / ((size_t)HWP * 128));
  const float a = 1.f - (float)tsel[b] / (float)TSTEPS;
  const float v = feat32[idx] - a * noise[idx];
  feat32[idx] = v;
  const int py = p >> 7, px = p & 127;
  xtB[((size_t)b * PHW + (py + 1) * PW + (px + 1)) * CHN + c] = f2bf(v);
}

// -------- pooled[b][c] = mean over HW of x_t -------------------------
__global__ void pool_k(const float* __restrict__ xt, float* __restrict__ pooled) {
  const int bc = blockIdx.x, b = bc >> 7, c = bc & 127;
  float s = 0.f;
  for (int p = threadIdx.x; p < HWP; p += 256) s += xt[((size_t)b * HWP + p) * CHN + c];
  __shared__ float ss[256];
  ss[threadIdx.x] = s; __syncthreads();
  for (int o = 128; o > 0; o >>= 1) { if (threadIdx.x < (unsigned)o) ss[threadIdx.x] += ss[threadIdx.x + o]; __syncthreads(); }
  if (threadIdx.x == 0) pooled[bc] = ss[0] / (float)HWP;
}

// -------- weight scalar: sigmoid(pooled . wg_w + wg_b) ---------------
__global__ void wg_k(const float* __restrict__ pooled, const float* __restrict__ wgw,
                     const float* __restrict__ wgb, float* __restrict__ out) {
  const int b = blockIdx.x, tid = threadIdx.x;
  __shared__ float red[128];
  red[tid] = pooled[b * 128 + tid] * wgw[tid];
  __syncthreads();
  for (int o = 64; o > 0; o >>= 1) { if (tid < o) red[tid] += red[tid + o]; __syncthreads(); }
  if (tid == 0) out[b] = sigf(red[0] + wgb[0]);
}

// -------- pad cr conv2 bias 3 -> 16 ----------------------------------
__global__ void padbias_k(const float* __restrict__ b3, float* __restrict__ b16) {
  const int i = threadIdx.x;
  if (i < 16) b16[i] = (i < 3) ? b3[i] : 0.f;
}

// =====================================================================
extern "C" void kernel_launch(void* const* d_in, const int* in_sizes, int n_in,
                              void* d_out, int out_size, void* d_ws, size_t ws_size,
                              hipStream_t stream) {
  (void)in_sizes; (void)n_in; (void)out_size; (void)ws_size;
  const float* x      = (const float*)d_in[0];
  const int*   t      = (const int*)  d_in[1];
  const float* fe_w1  = (const float*)d_in[2];
  const float* fe_b1  = (const float*)d_in[3];
  const float* fe_g1s = (const float*)d_in[4];
  const float* fe_g1b = (const float*)d_in[5];
  const float* fe_w2  = (const float*)d_in[6];
  const float* fe_b2  = (const float*)d_in[7];
  const float* fe_g2s = (const float*)d_in[8];
  const float* fe_g2b = (const float*)d_in[9];
  const float* mlp_w1 = (const float*)d_in[10];
  const float* mlp_b1 = (const float*)d_in[11];
  const float* mlp_w2 = (const float*)d_in[12];
  const float* mlp_b2 = (const float*)d_in[13];
  const float* c1w    = (const float*)d_in[14];
  const float* c1b    = (const float*)d_in[15];
  const float* g1s    = (const float*)d_in[16];
  const float* g1b    = (const float*)d_in[17];
  const float* c2w    = (const float*)d_in[18];
  const float* c2b    = (const float*)d_in[19];
  const float* g2s    = (const float*)d_in[20];
  const float* g2b    = (const float*)d_in[21];
  const float* c3w    = (const float*)d_in[22];
  const float* c3b    = (const float*)d_in[23];
  const float* cr_w1  = (const float*)d_in[24];
  const float* cr_b1  = (const float*)d_in[25];
  const float* cr_gs  = (const float*)d_in[26];
  const float* cr_gb  = (const float*)d_in[27];
  const float* cr_w2  = (const float*)d_in[28];
  const float* cr_b2  = (const float*)d_in[29];
  const float* wg_w   = (const float*)d_in[30];
  const float* wg_b   = (const float*)d_in[31];
  float* out = (float*)d_out;                 // [restored 8*3*16384][weight 8]

  // bump allocator on workspace
  char* ws = (char*)d_ws; size_t off = 0;
  auto alloc = [&](size_t bytes) -> void* {
    void* p = ws + off; off += (bytes + 255) & ~(size_t)255; return p;
  };
  float*          feat32 = (float*)alloc((size_t)BB * HWP * CHN * 4);   // 64 MB (also x_t)
  float*          tmp32  = (float*)alloc((size_t)BB * HWP * CHN * 4);   // 64 MB conv scratch
  unsigned short* bfA    = (unsigned short*)alloc((size_t)BB * PHW * CHN * 2); // padded
  unsigned short* bfB    = (unsigned short*)alloc((size_t)BB * PHW * CHN * 2); // padded
  unsigned short* wExp   = (unsigned short*)alloc((size_t)BB * 128 * 9 * 128 * 2);
  unsigned short* wFe2   = (unsigned short*)alloc((size_t)128 * 9 * 64 * 2);
  unsigned short* wCr1   = (unsigned short*)alloc((size_t)64 * 9 * 128 * 2);
  unsigned short* wCr2   = (unsigned short*)alloc((size_t)16 * 9 * 64 * 2);
  float*          bCr2   = (float*)alloc(16 * 4);
  float*          stats  = (float*)alloc(64 * 2 * 4);
  float*          tc     = (float*)alloc(BB * 4);
  float*          pooled = (float*)alloc(BB * 128 * 4);

  const dim3 cgrid(HWP / 16, BB);
  const unsigned bfN16 = (unsigned)((size_t)BB * PHW * CHN * 2 / 16);

  // zero padded bf16 buffers once per launch: halo stays zero throughout
  zero_u4_k<<<(bfN16 + 255) / 256, 256, 0, stream>>>((uint4*)bfA, bfN16);
  zero_u4_k<<<(bfN16 + 255) / 256, 256, 0, stream>>>((uint4*)bfB, bfN16);

  // feature extraction conv1 (3->64, VALU) + GN + SiLU -> bf16
  fe1_k<<<BB * HWP / 4, 256, 0, stream>>>(x, fe_w1, fe_b1, tmp32);
  gn_stats_k<<<64, 256, 0, stream>>>(tmp32, 64, 8, stats);
  gn_apply_k<<<BB * HWP * 64 / 256, 256, 0, stream>>>(tmp32, 64, 8, stats,
                                                      fe_g1s, fe_g1b, nullptr, 0, bfA, nullptr);
  // feature extraction conv2 (64->128, WMMA) + GN + SiLU -> feat (bf16 + fp32)
  stage_w_k<<<(128 * 9 * 64 + 255) / 256, 256, 0, stream>>>(fe_w2, wFe2, 128, 64, 128);
  conv3x3_wmma_k<64, 0><<<cgrid, 256, 0, stream>>>(bfA, wFe2, fe_b2, 0, nullptr,
                                                   tmp32, nullptr, 128);
  gn_stats_k<<<64, 256, 0, stream>>>(tmp32, 128, 16, stats);
  gn_apply_k<<<BB * HWP * 128 / 256, 256, 0, stream>>>(tmp32, 128, 16, stats,
                                                       fe_g2s, fe_g2b, nullptr, 0, bfB, feat32);
  // per-sample time MLP -> tc[b]
  mlp_tc_k<<<BB, 256, 0, stream>>>(t, mlp_w1, mlp_b1, mlp_w2, mlp_b2, tc);

  // expert conv1 (129->128): WMMA over first 128 cin + analytic time channel
  stage_we_k<<<BB * 128 * 9 * 128 / 256, 256, 0, stream>>>(c1w, t, wExp, 129);
  conv3x3_wmma_k<128, 0><<<cgrid, 256, 0, stream>>>(bfB, wExp, c1b, 128, t,
                                                    tmp32, nullptr, 128);
  tadd_k<<<BB * HWP * 128 / 256, 256, 0, stream>>>(tmp32, c1w, t, tc);
  gn_stats_k<<<64, 256, 0, stream>>>(tmp32, 128, 16, stats);
  gn_apply_k<<<BB * HWP * 128 / 256, 256, 0, stream>>>(tmp32, 128, 16, stats,
                                                       g1s, g1b, t, 128, bfA, nullptr);
  // expert conv2
  stage_we_k<<<BB * 128 * 9 * 128 / 256, 256, 0, stream>>>(c2w, t, wExp, 128);
  conv3x3_wmma_k<128, 0><<<cgrid, 256, 0, stream>>>(bfA, wExp, c2b, 128, t,
                                                    tmp32, nullptr, 128);
  gn_stats_k<<<64, 256, 0, stream>>>(tmp32, 128, 16, stats);
  gn_apply_k<<<BB * HWP * 128 / 256, 256, 0, stream>>>(tmp32, 128, 16, stats,
                                                       g2s, g2b, t, 128, bfB, nullptr);
  // expert conv3 -> noise
  stage_we_k<<<BB * 128 * 9 * 128 / 256, 256, 0, stream>>>(c3w, t, wExp, 128);
  conv3x3_wmma_k<128, 0><<<cgrid, 256, 0, stream>>>(bfB, wExp, c3b, 128, t,
                                                    tmp32, nullptr, 128);

  // x_t = feat - alpha*noise (fp32 in-place + bf16 padded), pooled, weight scalar
  xt_k<<<BB * HWP * 128 / 256, 256, 0, stream>>>(feat32, tmp32, t, bfA);
  pool_k<<<BB * 128, 256, 0, stream>>>(feat32, pooled);
  wg_k<<<BB, 128, 0, stream>>>(pooled, wg_w, wg_b, out + (size_t)BB * 3 * HWP);

  // color reconstruction conv1 (128->64, WMMA) + GN + SiLU
  stage_w_k<<<(64 * 9 * 128 + 255) / 256, 256, 0, stream>>>(cr_w1, wCr1, 64, 128, 64);
  conv3x3_wmma_k<128, 0><<<cgrid, 128, 0, stream>>>(bfA, wCr1, cr_b1, 0, nullptr,
                                                    tmp32, nullptr, 64);
  gn_stats_k<<<64, 256, 0, stream>>>(tmp32, 64, 8, stats);
  gn_apply_k<<<BB * HWP * 64 / 256, 256, 0, stream>>>(tmp32, 64, 8, stats,
                                                      cr_gs, cr_gb, nullptr, 0, bfB, nullptr);
  // color reconstruction conv2 (64->3 padded to 16, WMMA) + sigmoid -> restored
  stage_w_k<<<(16 * 9 * 64 + 255) / 256, 256, 0, stream>>>(cr_w2, wCr2, 3, 64, 16);
  padbias_k<<<1, 16, 0, stream>>>(cr_b2, bCr2);
  conv3x3_wmma_k<64, 1><<<cgrid, 32, 0, stream>>>(bfB, wCr2, bCr2, 0, nullptr,
                                                  nullptr, out, 16);
}
